// DiscMaker_45698452030018
// MI455X (gfx1250) — compile-verified, matrix operand order checked by reading, not run on hardware
//
#include <hip/hip_runtime.h>

// Problem constants
#define B_SZ 256
#define T_SZ 512
#define DIN  64
#define S_SZ 128
#define E_SZ 128
#define H_SZ 256

typedef __attribute__((ext_vector_type(16))) __bf16 v16bf;
typedef __attribute__((ext_vector_type(8)))  float  v8f;
typedef int vint4 __attribute__((__vector_size__(16)));

struct __attribute__((aligned(16))) U4 { unsigned int x, y, z, w; };
union V16 { v16bf v; unsigned int u[8]; U4 q[2]; };

// workspace byte offsets (packed bf16 weight tiles, then xproj)
#define OFF_W1   0u        //  64 KB : [Wx0|Wx1|Wx2|We]   K=64  N=512, 64 tiles
#define OFF_WHU  65536u    // 128 KB : [Wh0|Wh1|Wh2|Ue]   K=128 N=512, 128 tiles
#define OFF_WI   196608u   //  64 KB : Wi[0:128]          K=128 N=256, 64 tiles
#define OFF_WHC  262144u   // 128 KB : Whc                K=256 N=256, 128 tiles
#define OFF_WO   393216u   //   8 KB : Wo padded to N=16  K=256, 8 tiles
#define OFF_XP   401408u   // 128 MB : xproj bf16 [g][t][16][512]

__device__ __forceinline__ unsigned short f2bf(float f) {
  __bf16 h = (__bf16)f;
  return __builtin_bit_cast(unsigned short, h);
}
__device__ __forceinline__ float bf2f(unsigned short h) {
  unsigned int u = ((unsigned int)h) << 16;
  return __builtin_bit_cast(float, u);
}

// K index held by (lane, element e) of a 16-bit WMMA A/B operand (ISA 7.12.2)
__device__ __forceinline__ int kmapE(int lane, int e) {
  int vg = e >> 1, hf = e & 1;
  return ((vg & 4) ? 16 : 0) + ((lane & 16) ? 8 : 0) + ((vg & 3) * 2) + hf;
}

// Inverse map: flat position of element (row m, k) inside a packed-operand
// buffer laid out as [kb][lane][e] (1 KB per 32-K block).
__device__ __forceinline__ int apos(int m, int k) {
  int kb = k >> 5, kl = k & 31;
  int lane = m + ((kl & 8) ? 16 : 0);
  int vg = ((kl & 16) ? 4 : 0) + ((kl >> 1) & 3);
  return (kb << 9) + (lane << 4) + vg * 2 + (kl & 1);
}

__device__ __forceinline__ v8f wmma_bf(v16bf a, v16bf b, v8f c) {
  return __builtin_amdgcn_wmma_f32_16x16x32_bf16(false, a, false, b, (short)0, c,
                                                 false, false);
}

// Load one packed 16x32 operand (A or B): 32 bytes contiguous per lane.
__device__ __forceinline__ v16bf loadPK(const unsigned short* base, int tile,
                                        int lane) {
  V16 r;
  const U4* p = (const U4*)(base + ((size_t)tile << 9) + (lane << 4));
  r.q[0] = p[0];
  r.q[1] = p[1];
  return r.v;
}

// ---------------------------------------------------------------------------
// Phase 0: repack all weights (f32 row-major) into WMMA B-operand bf16 tiles.
// 392 tiles total, one wave per tile.
// ---------------------------------------------------------------------------
__global__ void rnn_pack_kernel(const float* __restrict__ Wx,
                                const float* __restrict__ Wh,
                                const float* __restrict__ We,
                                const float* __restrict__ Ue,
                                const float* __restrict__ Wi,
                                const float* __restrict__ Whc,
                                const float* __restrict__ Wo,
                                unsigned short* __restrict__ wsu) {
  int tile = blockIdx.x;
  int lane = threadIdx.x;  // 32 threads
  int nl = lane & 15;
#pragma unroll
  for (int e = 0; e < 16; ++e) {
    int kk = kmapE(lane, e);
    float val = 0.0f;
    size_t dst;
    if (tile < 64) {  // W1: K=64, N=512  ([Wx j | We])
      int loc = tile, nt = loc >> 1, kt = loc & 1;
      int k = kt * 32 + kk, n = nt * 16 + nl;
      val = (n < 384) ? Wx[((size_t)(n >> 7) * 64 + k) * 128 + (n & 127)]
                      : We[(size_t)k * 128 + (n - 384)];
      dst = (OFF_W1 >> 1) + (size_t)loc * 512;
    } else if (tile < 192) {  // WHU: K=128, N=512 ([Wh j | Ue])
      int loc = tile - 64, nt = loc >> 2, kt = loc & 3;
      int k = kt * 32 + kk, n = nt * 16 + nl;
      val = (n < 384) ? Wh[((size_t)(n >> 7) * 128 + k) * 128 + (n & 127)]
                      : Ue[(size_t)k * 128 + (n - 384)];
      dst = (OFF_WHU >> 1) + (size_t)loc * 512;
    } else if (tile < 256) {  // Wi[0:128]: K=128, N=256
      int loc = tile - 192, nt = loc >> 2, kt = loc & 3;
      int k = kt * 32 + kk, n = nt * 16 + nl;
      val = Wi[(size_t)k * 256 + n];
      dst = (OFF_WI >> 1) + (size_t)loc * 512;
    } else if (tile < 384) {  // Whc: K=256, N=256
      int loc = tile - 256, nt = loc >> 3, kt = loc & 7;
      int k = kt * 32 + kk, n = nt * 16 + nl;
      val = Whc[(size_t)k * 256 + n];
      dst = (OFF_WHC >> 1) + (size_t)loc * 512;
    } else {  // Wo: K=256, N=4 padded to 16
      int kt = tile - 384;
      int k = kt * 32 + kk;
      val = (nl < 4) ? Wo[(size_t)k * 4 + nl] : 0.0f;
      dst = (OFF_WO >> 1) + (size_t)kt * 512;
    }
    wsu[dst + (size_t)lane * 16 + e] = f2bf(val);
  }
}

// ---------------------------------------------------------------------------
// Phase 1: xproj[g][t][m][0:512] = [ xt@Wx0 | xt@Wx1 | xt@Wx2 | xt@We ]  (bf16)
// Fully parallel GEMM over all (b,t). grid = 16*512 blocks of 8 waves;
// each wave does 4 N-tiles of 16 via 2 WMMAs (K=64).
// ---------------------------------------------------------------------------
__global__ void rnn_xproj_kernel(const float* __restrict__ x,
                                 const unsigned short* __restrict__ wsu,
                                 unsigned short* __restrict__ xproj) {
  int bid = blockIdx.x;
  int g = bid >> 9;
  int t = bid & 511;
  int lane = threadIdx.x & 31;
  int wv = threadIdx.x >> 5;

  const unsigned short* w1 = wsu + (OFF_W1 >> 1);

  // A operands: 16 rows (global batch g*16+m) of xt, K=0..63
  V16 a[2];
  int b0 = g * 16 + (lane & 15);
  const float* xr = x + ((size_t)b0 * T_SZ + t) * DIN;
#pragma unroll
  for (int kb = 0; kb < 2; ++kb) {
#pragma unroll
    for (int p = 0; p < 8; ++p) {
      int K = kb * 32 + kmapE(lane, 2 * p);
      float f0 = xr[K];
      float f1 = xr[K + 1];
      a[kb].u[p] = (unsigned int)f2bf(f0) | ((unsigned int)f2bf(f1) << 16);
    }
  }

  size_t obase = ((size_t)(g * T_SZ + t)) * 16 * 512;
#pragma unroll
  for (int i = 0; i < 4; ++i) {
    int nt = wv * 4 + i;
    v8f acc = {0.f, 0.f, 0.f, 0.f, 0.f, 0.f, 0.f, 0.f};
#pragma unroll
    for (int kb = 0; kb < 2; ++kb) {
      v16bf bb = loadPK(w1, nt * 2 + kb, lane);
      acc = wmma_bf(a[kb].v, bb, acc);
    }
    int n = lane & 15;
    int mb = (lane >> 4) << 3;
#pragma unroll
    for (int v = 0; v < 8; ++v) {
      int m = v + mb;
      xproj[obase + (size_t)m * 512 + nt * 16 + n] = f2bf(acc[v]);
    }
  }
}

// ---------------------------------------------------------------------------
// Phase 2: persistent recurrent kernel. 16 WGs (one per 16-row batch tile),
// 8 waves each, 512 sequential steps. Whc cached in LDS (async load-to-LDS
// when available); activation staging buffers stored directly in WMMA
// A-operand layout so every operand load is 2x ds_load_b128.
// ---------------------------------------------------------------------------
__global__ void rnn_recurrent_kernel(const float* __restrict__ y,
                                     const float* __restrict__ WiFull,
                                     const float* __restrict__ Wd,
                                     const float* __restrict__ bd,
                                     const float* __restrict__ gate0,
                                     const unsigned short* __restrict__ wsu,
                                     float* __restrict__ out) {
  extern __shared__ unsigned char smem[];
  unsigned short* whcL  = (unsigned short*)smem;            // 131072 B
  unsigned short* kstP  = (unsigned short*)(smem + 131072); //   4096 B (16xK128 packed)
  unsigned short* encP  = (unsigned short*)(smem + 135168); //   4096 B (16xK128 packed)
  unsigned short* csb0  = (unsigned short*)(smem + 139264); //   8192 B (16xK256 packed)
  unsigned short* csb1  = (unsigned short*)(smem + 147456); //   8192 B
  float* nst            = (float*)(smem + 155648);          //  24576 B (3x16x128)
  float* out16          = (float*)(smem + 180224);          //   1024 B (16x16)
  float* gate_s         = (float*)(smem + 181248);          //    192 B (16x3)
  float* err_s          = (float*)(smem + 181440);          //     64 B

  int g = blockIdx.x;
  int tid = threadIdx.x;
  int lane = tid & 31;
  int wv = tid >> 5;

  const unsigned short* whu = wsu + (OFF_WHU >> 1);
  const unsigned short* wiP = wsu + (OFF_WI >> 1);
  const unsigned short* whcG = wsu + (OFF_WHC >> 1);
  const unsigned short* woP = wsu + (OFF_WO >> 1);
  const unsigned short* xpA = wsu + (OFF_XP >> 1);

  // ---- init: zero states, load gate0, cache Whc tiles in LDS ----
  for (int i = tid; i < 1024; i += 256) ((unsigned int*)kstP)[i] = 0u;
  for (int i = tid; i < 2048; i += 256) ((unsigned int*)csb0)[i] = 0u;
  if (tid < 16) err_s[tid] = 0.0f;
  for (int i = tid; i < 48; i += 256) gate_s[i] = gate0[(size_t)g * 48 + i];

#if defined(__HIP_DEVICE_COMPILE__) && \
    __has_builtin(__builtin_amdgcn_global_load_async_to_lds_b128)
  {
    // ASYNCcnt-tracked bulk copy: global (L2) -> LDS, 16 B per lane per issue.
    // Builtin signature (from compiler diagnostic): param 1 is
    // `int __vector(4) __device__ *` (AS1), param 2 the AS3 counterpart.
    __attribute__((address_space(1))) vint4* src =
        (__attribute__((address_space(1))) vint4*)(vint4*)(unsigned short*)whcG;
    __attribute__((address_space(3))) vint4* dst =
        (__attribute__((address_space(3))) vint4*)(vint4*)whcL;
    for (int i = tid; i < 8192; i += 256)
      __builtin_amdgcn_global_load_async_to_lds_b128(src + i, dst + i, 0, 0);
#if __has_builtin(__builtin_amdgcn_s_wait_asynccnt)
    __builtin_amdgcn_s_wait_asynccnt(0);
#endif
  }
#else
  {
    const U4* src = (const U4*)whcG;
    U4* dst = (U4*)whcL;
    for (int i = tid; i < 8192; i += 256) dst[i] = src[i];
  }
#endif
  __syncthreads();

  for (int t = 0; t < T_SZ; ++t) {
    const unsigned short* xp = xpA + (size_t)(g * T_SZ + t) * 16 * 512;
    const unsigned short* csR = (t & 1) ? csb1 : csb0;
    unsigned short* csW = (t & 1) ? csb0 : csb1;
    int n = lane & 15;
    int mb = (lane >> 4) << 3;

    // Pull next step's xproj block (16 KB) toward the WGP while computing.
    if (t + 1 < T_SZ)
      __builtin_prefetch(xp + 16 * 512 + (size_t)tid * 32, 0, 1);

    // ---- part 1: new_state (3x) and encoded  (A = kstate, K=128) ----
    {
      V16 a[4];
#pragma unroll
      for (int kb = 0; kb < 4; ++kb) a[kb].v = loadPK(kstP, kb, lane);
#pragma unroll
      for (int i = 0; i < 4; ++i) {
        int nt = wv * 4 + i;  // 0..31: [0,24)=new_state, [24,32)=encoded
        v8f acc = {0.f, 0.f, 0.f, 0.f, 0.f, 0.f, 0.f, 0.f};
#pragma unroll
        for (int kb = 0; kb < 4; ++kb) {
          v16bf bb = loadPK(whu, nt * 4 + kb, lane);
          acc = wmma_bf(a[kb].v, bb, acc);
        }
#pragma unroll
        for (int v = 0; v < 8; ++v) {
          int m = v + mb;
          float xv = bf2f(xp[(size_t)m * 512 + nt * 16 + n]);
          float tv = tanhf(acc[v] + xv);
          if (nt < 24) {
            int j = nt >> 3, s = (nt & 7) * 16 + n;
            nst[((size_t)j * 16 + m) * S_SZ + s] = tv;
          } else {
            encP[apos(m, (nt - 24) * 16 + n)] = f2bf(tv);
          }
        }
      }
    }
    __syncthreads();

    // ---- part 2: cstate = tanh(enc@Wi + err*Wi_last + cstate@Whc) ----
    {
      V16 ae[4], ac[8];
#pragma unroll
      for (int kb = 0; kb < 4; ++kb) ae[kb].v = loadPK(encP, kb, lane);
#pragma unroll
      for (int kb = 0; kb < 8; ++kb) ac[kb].v = loadPK(csR, kb, lane);
#pragma unroll
      for (int i = 0; i < 2; ++i) {
        int nt = wv * 2 + i;  // 0..15
        v8f acc = {0.f, 0.f, 0.f, 0.f, 0.f, 0.f, 0.f, 0.f};
#pragma unroll
        for (int kb = 0; kb < 4; ++kb) {
          v16bf bb = loadPK(wiP, nt * 4 + kb, lane);
          acc = wmma_bf(ae[kb].v, bb, acc);
        }
#pragma unroll
        for (int kb = 0; kb < 8; ++kb) {
          v16bf bb = loadPK(whcL, nt * 8 + kb, lane);
          acc = wmma_bf(ac[kb].v, bb, acc);
        }
        int col = nt * 16 + n;
        float wl = WiFull[(size_t)128 * H_SZ + col];  // Wi last row (err input)
#pragma unroll
        for (int v = 0; v < 8; ++v) {
          int m = v + mb;
          float cv = tanhf(acc[v] + err_s[m] * wl);
          csW[apos(m, col)] = f2bf(cv);
        }
      }
    }
    __syncthreads();

    // ---- part 3+4 (wave 0): out = cstate@Wo (padded N=16); gate math ----
    if (wv == 0) {
      v8f acc = {0.f, 0.f, 0.f, 0.f, 0.f, 0.f, 0.f, 0.f};
#pragma unroll
      for (int kb = 0; kb < 8; ++kb) {
        v16bf aa = loadPK(csW, kb, lane);
        v16bf bb = loadPK(woP, kb, lane);
        acc = wmma_bf(aa, bb, acc);
      }
#pragma unroll
      for (int v = 0; v < 8; ++v) out16[(size_t)(v + mb) * 16 + n] = acc[v];

      if (lane < 16) {
        int b = lane;
        float o0 = out16[b * 16 + 0], o1 = out16[b * 16 + 1];
        float o2 = out16[b * 16 + 2], o3 = out16[b * 16 + 3];
        float d0 = o0 * Wd[0] + o1 * Wd[3] + o2 * Wd[6] + bd[0];
        float d1 = o0 * Wd[1] + o1 * Wd[4] + o2 * Wd[7] + bd[1];
        float d2 = o0 * Wd[2] + o1 * Wd[5] + o2 * Wd[8] + bd[2];
        float mx = fmaxf(d0, fmaxf(d1, d2));
        float e0 = expf(d0 - mx), e1 = expf(d1 - mx), e2 = expf(d2 - mx);
        float inv = 1.0f / (e0 + e1 + e2);
        float th = 1.0f / (1.0f + expf(-o3));
        float om = 1.0f - th;
        gate_s[b * 3 + 0] = e0 * inv * th + gate_s[b * 3 + 0] * om;
        gate_s[b * 3 + 1] = e1 * inv * th + gate_s[b * 3 + 1] * om;
        gate_s[b * 3 + 2] = e2 * inv * th + gate_s[b * 3 + 2] * om;
      }
    }
    __syncthreads();

    // ---- part 5: kstate = sum_j gate_j * new_state_j ; pred, err ----
#pragma unroll
    for (int q = 0; q < 8; ++q) {
      int flat = tid * 8 + q;  // 16*128 = 2048 elements
      int b = flat >> 7;
      int s = flat & 127;
      float g0 = gate_s[b * 3 + 0], g1 = gate_s[b * 3 + 1], g2 = gate_s[b * 3 + 2];
      float kv = g0 * nst[(size_t)b * S_SZ + s] +
                 g1 * nst[(size_t)(16 + b) * S_SZ + s] +
                 g2 * nst[(size_t)(32 + b) * S_SZ + s];
      kstP[apos(b, s)] = f2bf(kv);
      if (s == 127) {
        int gb = g * 16 + b;
        out[(size_t)gb * T_SZ + t] = kv;  // pred (f32)
        err_s[b] = kv - y[(size_t)gb * T_SZ + t];
      }
    }
    __syncthreads();
  }
}

extern "C" void kernel_launch(void* const* d_in, const int* in_sizes, int n_in,
                              void* d_out, int out_size, void* d_ws, size_t ws_size,
                              hipStream_t stream) {
  (void)in_sizes; (void)n_in; (void)out_size; (void)ws_size;
  const float* x     = (const float*)d_in[0];
  const float* y     = (const float*)d_in[1];
  const float* Wx    = (const float*)d_in[2];
  const float* Wh    = (const float*)d_in[3];
  const float* We    = (const float*)d_in[4];
  const float* Ue    = (const float*)d_in[5];
  const float* Wi    = (const float*)d_in[6];
  const float* Whc   = (const float*)d_in[7];
  const float* Wo    = (const float*)d_in[8];
  const float* Wd    = (const float*)d_in[9];
  const float* bd    = (const float*)d_in[10];
  const float* gate0 = (const float*)d_in[11];
  unsigned short* wsu = (unsigned short*)d_ws;

  // Phase 0: pack weights into WMMA bf16 tile layout (392 tiles).
  rnn_pack_kernel<<<392, 32, 0, stream>>>(Wx, Wh, We, Ue, Wi, Whc, Wo, wsu);

  // Phase 1: hoisted input projections (parallel WMMA GEMM over all t).
  rnn_xproj_kernel<<<16 * T_SZ, 256, 0, stream>>>(x, wsu, wsu + (OFF_XP >> 1));

  // Phase 2: persistent recurrent scan, 16 independent batch tiles.
  size_t smem_bytes = 181504;
  rnn_recurrent_kernel<<<16, 256, smem_bytes, stream>>>(y, Wi, Wd, bd, gate0,
                                                        wsu, (float*)d_out);
}
